// SelfAttentionWithCache_84963043049653
// MI455X (gfx1250) — matrix operand order repaired and checked by reading, not compiled, not run on hardware
//
#include <hip/hip_runtime.h>
#include <stdint.h>

#define D_MODEL 2048
#define N_HEADS 16
#define HEAD_DIM 128
#define BATCH 2
#define SEQ 2048
#define PASTN 2048
#define FULLS (PASTN + SEQ)     // 4096
#define MROWS (BATCH * SEQ)     // 4096

typedef __bf16 v16bf __attribute__((ext_vector_type(16)));
typedef float  v8f   __attribute__((ext_vector_type(8)));
typedef unsigned int u32x4 __attribute__((ext_vector_type(4)));
typedef int          i32x8 __attribute__((ext_vector_type(8)));
typedef int          i32x4 __attribute__((ext_vector_type(4)));

#if defined(__has_builtin)
#if __has_builtin(__builtin_amdgcn_tensor_load_to_lds) && \
    __has_builtin(__builtin_amdgcn_s_wait_tensorcnt)
#define USE_TDM 1
#endif
#endif
#ifndef USE_TDM
#define USE_TDM 0
#endif

static __device__ __forceinline__ uint16_t f2bf(float f) {
  union { float f; uint32_t u; } x; x.f = f;
  uint32_t r = (x.u + 0x7FFFu + ((x.u >> 16) & 1u)) >> 16;
  return (uint16_t)r;
}

static __device__ __forceinline__ v8f vzero8() {
  v8f z = {0.f,0.f,0.f,0.f,0.f,0.f,0.f,0.f};
  return z;
}

// A-fragment (16x32 bf16, M x K): lane L holds row m=L%16;
// VGPR0-3 = K[kb..kb+7], VGPR4-7 = K[kb+16..kb+23], kb = 8*(L/16).
static __device__ __forceinline__ v16bf ldfragA(const uint16_t* base, int lda,
                                                int m0, int k0, int lane) {
  int m  = m0 + (lane & 15);
  int kb = k0 + ((lane >> 4) << 3);
  const uint16_t* p = base + (size_t)m * (size_t)lda + kb;
  union { uint4 q[2]; v16bf v; } u;
  u.q[0] = *reinterpret_cast<const uint4*>(p);
  u.q[1] = *reinterpret_cast<const uint4*>(p + 16);
  return u.v;
}

// B-fragment (32x16 bf16, K x N) from a row-major [N x K] matrix (B[k][n] = M[n][k]):
// lane L holds col n=L%16; VGPR0-3 = K[kb..kb+7], VGPR4-7 = K[kb+16..kb+23], kb = 8*(L/16).
static __device__ __forceinline__ v16bf ldfragB(const uint16_t* base, int ldb,
                                                int n0, int k0, int lane) {
  int n  = n0 + (lane & 15);
  int kb = k0 + ((lane >> 4) << 3);
  const uint16_t* p = base + (size_t)n * (size_t)ldb + kb;
  union { uint4 q[2]; v16bf v; } u;
  u.q[0] = *reinterpret_cast<const uint4*>(p);
  u.q[1] = *reinterpret_cast<const uint4*>(p + 16);
  return u.v;
}

#if USE_TDM
// TDM DMA: load a 2-D tile (tile_d0 contiguous elems x tile_d1 rows, 2-byte
// elements) from a row-major tensor (row stride = stride0 elems) into LDS,
// packed contiguously (ldb = tile_d0). D# layout per CDNA5 ISA 8.3/8.4.
// This toolchain (clang-23 / therock-10.0) exposes the 6-arg builtin:
// (u32x4 g0, i32x8 g1, i32x4 g2, i32x4 g3, i32x8 extra, i32 cpol).
static __device__ __forceinline__ void tdm_load_2d_bf16(
    const uint16_t* gptr, uint32_t lds_off_bytes,
    uint32_t tensor_d0, uint32_t tensor_d1, uint32_t stride0,
    uint32_t tile_d0, uint32_t tile_d1) {
  uint64_t ga = (uint64_t)(uintptr_t)gptr;
  u32x4 g0;
  g0[0] = 1u;                                   // count=1, user descriptor
  g0[1] = lds_off_bytes;                        // lds_addr [63:32]
  g0[2] = (uint32_t)ga;                         // global_addr [95:64]
  g0[3] = (uint32_t)((ga >> 32) & 0x01FFFFFFu)  // global_addr [120:96]
        | (2u << 30);                           // type=2 (image)
  i32x8 g1;
  g1[0] = (int)(1u << 16);                      // wg_mask=0, data_size=1 (2B)
  g1[1] = (int)((tensor_d0 & 0xFFFFu) << 16);   // abar_addr=0 | td0.lo16
  g1[2] = (int)((tensor_d0 >> 16) | ((tensor_d1 & 0xFFFFu) << 16));
  g1[3] = (int)((tensor_d1 >> 16) | (tile_d0 << 16));
  g1[4] = (int)(tile_d1 & 0xFFFFu);             // tile_dim1 | tile_dim2=0
  g1[5] = (int)stride0;                         // tensor_dim0_stride lo32
  g1[6] = 0;                                    // stride0 hi16 | stride1 lo16
  g1[7] = 0;                                    // stride1 hi32
  i32x4 z4 = {0, 0, 0, 0};                      // groups 2/3: dims 3/4 unused
  i32x8 z8 = {0, 0, 0, 0, 0, 0, 0, 0};
  __builtin_amdgcn_tensor_load_to_lds(g0, g1, z4, z4, z8, 0);
}
#endif

// ---------------------------------------------------------------------------
// elementwise f32 -> bf16 convert
__global__ void convert_bf16_kernel(const float* __restrict__ src,
                                    uint16_t* __restrict__ dst, int n) {
  int i = blockIdx.x * 256 + threadIdx.x;
  if (i < n) dst[i] = f2bf(src[i]);
}

// past_k/past_v -> present (f32) + bf16 caches (kb row-major, vbT transposed)
__global__ void prep_past_kernel(const float* __restrict__ pk,
                                 const float* __restrict__ pv,
                                 float* __restrict__ presK,
                                 float* __restrict__ presV,
                                 uint16_t* __restrict__ kbb,
                                 uint16_t* __restrict__ vbT) {
  int idx = blockIdx.x * 256 + threadIdx.x;          // [0, B*H*PAST*HD)
  int d  = idx & 127;
  int p  = (idx >> 7) & 2047;
  int bh = idx >> 18;                                 // b*16+h
  size_t di = (((size_t)bh) * FULLS + p) * HEAD_DIM + d;
  float kv = pk[idx];
  presK[di] = kv;
  kbb[di]   = f2bf(kv);
  float vv = pv[idx];
  presV[di] = vv;
  vbT[(((size_t)bh) * HEAD_DIM + d) * FULLS + p] = f2bf(vv);
}

// ---------------------------------------------------------------------------
// C[M,N] = A[M,K] (bf16, row-major) x B[N,K]^T (bf16, row-major) via WMMA bf16.
// The 128x32 B tile for the block is staged in LDS by the Tensor Data Mover
// (double-buffered); A fragments come straight from global (L2-resident).
// MODE 0: outF[m*N+n] = c                       (final projection, f32)
// MODE 1: outB[m*N+n] = bf16(c)                 (Q buffer)
// MODE 2: present_k f32 + kb bf16, head layout  (K projection)
// MODE 3: present_v f32 + vbT bf16 transposed   (V projection)
template <int MODE>
__global__ __launch_bounds__(128)
void gemm_wmma_kernel(const uint16_t* __restrict__ A,
                      const uint16_t* __restrict__ B,
                      float* __restrict__ outF,
                      uint16_t* __restrict__ outB,
                      int M, int N, int K) {
  __shared__ __align__(16) uint16_t tileB[2][128 * 32];  // 2 x 8KB
  int lane = threadIdx.x & 31;
  int w    = threadIdx.x >> 5;
  int m0   = blockIdx.y * 128 + (w & 1) * 64;
  int nblk = blockIdx.x * 128;
  int n0l  = (w >> 1) * 64;  // n offset within the block tile

  v8f acc[4][4];
#pragma unroll
  for (int i = 0; i < 4; ++i)
#pragma unroll
    for (int j = 0; j < 4; ++j) acc[i][j] = vzero8();

#if USE_TDM
  if (w == 0)
    tdm_load_2d_bf16(B + (size_t)nblk * K, (uint32_t)(uintptr_t)&tileB[0][0],
                     (uint32_t)K, (uint32_t)N, (uint32_t)K, 32u, 128u);
#endif

  int cur = 0;
  for (int k0 = 0; k0 < K; k0 += 32, cur ^= 1) {
#if USE_TDM
    if (w == 0) {
      if (k0 + 32 < K) {
        tdm_load_2d_bf16(B + (size_t)nblk * K + (k0 + 32),
                         (uint32_t)(uintptr_t)&tileB[cur ^ 1][0],
                         (uint32_t)K, (uint32_t)N, (uint32_t)K, 32u, 128u);
        __builtin_amdgcn_s_wait_tensorcnt(1);
      } else {
        __builtin_amdgcn_s_wait_tensorcnt(0);
      }
    }
    __syncthreads();
#endif
    v16bf af[4], bf[4];
#pragma unroll
    for (int i = 0; i < 4; ++i) af[i] = ldfragA(A, K, m0 + 16 * i, k0, lane);
#if USE_TDM
#pragma unroll
    for (int j = 0; j < 4; ++j)
      bf[j] = ldfragB(&tileB[cur][0], 32, n0l + 16 * j, 0, lane);
#else
#pragma unroll
    for (int j = 0; j < 4; ++j)
      bf[j] = ldfragB(B, K, nblk + n0l + 16 * j, k0, lane);
#endif
    if (k0 + 64 < K)
      __builtin_prefetch(A + (size_t)(m0 + (lane & 15)) * K + k0 + 64, 0, 1);
#pragma unroll
    for (int i = 0; i < 4; ++i)
#pragma unroll
      for (int j = 0; j < 4; ++j)
        acc[i][j] = __builtin_amdgcn_wmma_f32_16x16x32_bf16(
            false, af[i], false, bf[j], (short)0, acc[i][j], false, false);
#if USE_TDM
    __syncthreads();  // all waves done with tileB[cur] before TDM reuses it
#endif
  }

  int hi = lane >> 4, col = lane & 15;
#pragma unroll
  for (int i = 0; i < 4; ++i)
#pragma unroll
    for (int j = 0; j < 4; ++j)
#pragma unroll
      for (int r = 0; r < 8; ++r) {
        int m = m0 + 16 * i + r + 8 * hi;
        int n = nblk + n0l + 16 * j + col;
        float c = acc[i][j][r];
        if (MODE == 0) {
          outF[(size_t)m * N + n] = c;
        } else if (MODE == 1) {
          outB[(size_t)m * N + n] = f2bf(c);
        } else {
          int b = m >> 11, s = m & 2047;
          int h = n >> 7, d = n & 127;
          size_t ci = (((size_t)(b * N_HEADS + h)) * FULLS + (PASTN + s)) * HEAD_DIM + d;
          outF[ci] = c;
          if (MODE == 2)
            outB[ci] = f2bf(c);
          else
            outB[(((size_t)(b * N_HEADS + h)) * HEAD_DIM + d) * FULLS + (PASTN + s)] =
                f2bf(c);
        }
      }
}

// ---------------------------------------------------------------------------
// Flash attention: one wave = 16 query rows of one (b,h); streams keys in
// 32-wide tiles with online softmax; P re-laid out via LDS; O kept in regs.
__global__ __launch_bounds__(128)
void attn_fwd_kernel(const uint16_t* __restrict__ qb,
                     const uint16_t* __restrict__ kbb,
                     const uint16_t* __restrict__ vbT,
                     uint16_t* __restrict__ attnb) {
  __shared__ __align__(16) uint16_t plds[4][16 * 32];
  int lane = threadIdx.x & 31;
  int w    = threadIdx.x >> 5;
  int b = blockIdx.z, h = blockIdx.y;
  int qbase = blockIdx.x * 64 + w * 16;
  int hi = lane >> 4, col = lane & 15;

  const uint16_t* kbh = kbb + (((size_t)(b * N_HEADS + h)) * FULLS) * HEAD_DIM;
  const uint16_t* vbh = vbT + (((size_t)(b * N_HEADS + h)) * HEAD_DIM) * FULLS;

  v16bf qf[4];
#pragma unroll
  for (int t = 0; t < 4; ++t)
    qf[t] = ldfragA(qb, D_MODEL, b * SEQ + qbase, h * HEAD_DIM + 32 * t, lane);

  v8f o[8];
#pragma unroll
  for (int dt = 0; dt < 8; ++dt) o[dt] = vzero8();
  float mrow[8], lrow[8];
#pragma unroll
  for (int r = 0; r < 8; ++r) { mrow[r] = -__builtin_inff(); lrow[r] = 0.f; }

  const float SM = 0.08838834764831845f;  // 1/sqrt(128)
  int kend = PASTN + qbase + 16;          // last visible key (exclusive)

  for (int kk0 = 0; kk0 < kend; kk0 += 32) {
    v8f s0 = vzero8(), s1 = vzero8();
#pragma unroll
    for (int t = 0; t < 4; ++t) {
      v16bf kf0 = ldfragB(kbh, HEAD_DIM, kk0, 32 * t, lane);
      v16bf kf1 = ldfragB(kbh, HEAD_DIM, kk0 + 16, 32 * t, lane);
      s0 = __builtin_amdgcn_wmma_f32_16x16x32_bf16(false, qf[t], false, kf0,
                                                   (short)0, s0, false, false);
      s1 = __builtin_amdgcn_wmma_f32_16x16x32_bf16(false, qf[t], false, kf1,
                                                   (short)0, s1, false, false);
    }
    bool needMask = (kk0 + 31 > PASTN + qbase);
    float scr[8];
#pragma unroll
    for (int r = 0; r < 8; ++r) {
      float a = s0[r] * SM;
      float c = s1[r] * SM;
      if (needMask) {
        int lim = PASTN + qbase + r + 8 * hi;  // max visible key index
        if (kk0 + col > lim)      a = -__builtin_inff();
        if (kk0 + 16 + col > lim) c = -__builtin_inff();
      }
      float mx = fmaxf(a, c);
#pragma unroll
      for (int sft = 1; sft < 16; sft <<= 1) mx = fmaxf(mx, __shfl_xor(mx, sft, 32));
      float nm = fmaxf(mrow[r], mx);
      float sc = __expf(mrow[r] - nm);
      mrow[r] = nm;
      float p0 = __expf(a - nm);
      float p1 = __expf(c - nm);
      float rs = p0 + p1;
#pragma unroll
      for (int sft = 1; sft < 16; sft <<= 1) rs += __shfl_xor(rs, sft, 32);
      lrow[r] = lrow[r] * sc + rs;
      scr[r] = sc;
      int m = r + 8 * hi;
      plds[w][m * 32 + col]      = f2bf(p0);
      plds[w][m * 32 + 16 + col] = f2bf(p1);
    }
#pragma unroll
    for (int dt = 0; dt < 8; ++dt)
#pragma unroll
      for (int r = 0; r < 8; ++r) o[dt][r] *= scr[r];

    asm volatile("s_wait_dscnt 0" ::: "memory");
    v16bf pf = ldfragA(&plds[w][0], 32, 0, 0, lane);
#pragma unroll
    for (int dt = 0; dt < 8; ++dt) {
      v16bf vf = ldfragB(vbh, FULLS, dt * 16, kk0, lane);
      o[dt] = __builtin_amdgcn_wmma_f32_16x16x32_bf16(false, pf, false, vf,
                                                      (short)0, o[dt], false, false);
    }
  }

#pragma unroll
  for (int dt = 0; dt < 8; ++dt)
#pragma unroll
    for (int r = 0; r < 8; ++r) {
      int m = r + 8 * hi;
      float v = o[dt][r] / lrow[r];
      size_t row = (size_t)(b * SEQ + qbase + m);
      attnb[row * D_MODEL + h * HEAD_DIM + dt * 16 + col] = f2bf(v);
    }
}

// ---------------------------------------------------------------------------
extern "C" void kernel_launch(void* const* d_in, const int* in_sizes, int n_in,
                              void* d_out, int out_size, void* d_ws, size_t ws_size,
                              hipStream_t stream) {
  const float* x      = (const float*)d_in[0];
  const float* past_k = (const float*)d_in[1];
  const float* past_v = (const float*)d_in[2];
  const float* Wq     = (const float*)d_in[3];
  const float* Wk     = (const float*)d_in[4];
  const float* Wv     = (const float*)d_in[5];
  const float* Wo     = (const float*)d_in[6];

  float* out   = (float*)d_out;                                   // [B,S,D]
  float* presK = out + (size_t)MROWS * D_MODEL;                   // [B,H,FULL,HD]
  float* presV = presK + (size_t)BATCH * N_HEADS * FULLS * HEAD_DIM;

  char* ws = (char*)d_ws;
  size_t off = 0;
  uint16_t* xb   = (uint16_t*)(ws + off); off += (size_t)MROWS * D_MODEL * 2;
  uint16_t* wqb  = (uint16_t*)(ws + off); off += (size_t)D_MODEL * D_MODEL * 2;
  uint16_t* wkb  = (uint16_t*)(ws + off); off += (size_t)D_MODEL * D_MODEL * 2;
  uint16_t* wvb  = (uint16_t*)(ws + off); off += (size_t)D_MODEL * D_MODEL * 2;
  uint16_t* wob  = (uint16_t*)(ws + off); off += (size_t)D_MODEL * D_MODEL * 2;
  uint16_t* qbuf = (uint16_t*)(ws + off); off += (size_t)MROWS * D_MODEL * 2;
  uint16_t* kbb  = (uint16_t*)(ws + off); off += (size_t)BATCH * N_HEADS * FULLS * HEAD_DIM * 2;
  uint16_t* vbT  = (uint16_t*)(ws + off); off += (size_t)BATCH * N_HEADS * FULLS * HEAD_DIM * 2;
  uint16_t* attnb= (uint16_t*)(ws + off); off += (size_t)MROWS * D_MODEL * 2;

  int nx = MROWS * D_MODEL;        // 8,388,608
  int nw = D_MODEL * D_MODEL;      // 4,194,304
  convert_bf16_kernel<<<nx / 256, 256, 0, stream>>>(x,  xb,  nx);
  convert_bf16_kernel<<<nw / 256, 256, 0, stream>>>(Wq, wqb, nw);
  convert_bf16_kernel<<<nw / 256, 256, 0, stream>>>(Wk, wkb, nw);
  convert_bf16_kernel<<<nw / 256, 256, 0, stream>>>(Wv, wvb, nw);
  convert_bf16_kernel<<<nw / 256, 256, 0, stream>>>(Wo, wob, nw);

  int np = BATCH * N_HEADS * PASTN * HEAD_DIM;  // 8,388,608
  prep_past_kernel<<<np / 256, 256, 0, stream>>>(past_k, past_v, presK, presV, kbb, vbT);

  dim3 ggrid(D_MODEL / 128, MROWS / 128);
  gemm_wmma_kernel<1><<<ggrid, 128, 0, stream>>>(xb, wqb, nullptr, qbuf,
                                                 MROWS, D_MODEL, D_MODEL);
  gemm_wmma_kernel<2><<<ggrid, 128, 0, stream>>>(xb, wkb, presK, kbb,
                                                 MROWS, D_MODEL, D_MODEL);
  gemm_wmma_kernel<3><<<ggrid, 128, 0, stream>>>(xb, wvb, presV, vbT,
                                                 MROWS, D_MODEL, D_MODEL);

  attn_fwd_kernel<<<dim3(SEQ / 64, N_HEADS, BATCH), 128, 0, stream>>>(qbuf, kbb, vbT, attnb);

  gemm_wmma_kernel<0><<<ggrid, 128, 0, stream>>>(attnb, wob, out, nullptr,
                                                 MROWS, D_MODEL, D_MODEL);
}
